// SwitchTransformersSparseMLP_69965017252081
// MI455X (gfx1250) — compile-verified, matrix operand order checked by reading, not run on hardware
//
#include <hip/hip_runtime.h>
#include <hip/hip_bf16.h>
#include <stdint.h>

#define DM   768
#define DF   3072
#define NE   8
#define NTOK 4096
#define LSTR 40   // LDS row stride in bf16 elements (80B): conflict-free frag reads

typedef __attribute__((ext_vector_type(16))) __bf16 v16bf;
typedef __attribute__((ext_vector_type(8)))  float  v8f;

// ---------- helpers ----------
__device__ __forceinline__ unsigned short f2bf(float f) {
  union { __bf16 b; unsigned short u; } c;
  c.b = (__bf16)f;                    // native v_cvt_bf16_f32 (RNE)
  return c.u;
}

__device__ __forceinline__ unsigned pack2(float lo, float hi) {
  return (unsigned)f2bf(lo) | ((unsigned)f2bf(hi) << 16);
}

// Load a 16x32 bf16 WMMA operand fragment from a row-major LDS tile
// (rows = M or N, cols = K). ISA layout: lane L -> row L&15;
// VGPR i holds K pair at (i&3)*2 + (i>>2)*16 + 8*(L>=16).
__device__ __forceinline__ v16bf load_frag(const unsigned short* base, int lane) {
  const int r  = lane & 15;
  const int kh = (lane >> 4) << 3;
  const unsigned short* p = base + r * LSTR;
  union { v16bf v; unsigned u[8]; } f;
#pragma unroll
  for (int i = 0; i < 8; ++i) {
    const int kb = (i & 3) * 2 + (i >> 2) * 16 + kh;
    f.u[i] = *(const unsigned*)(p + kb);     // 2 packed bf16, 4B aligned (kb even)
  }
  return f.v;
}

// Stage a 32(K) x 128(N) fp32 tile transposed into LDS as bf16 [n][k].
// Wave w handles K rows 4w..4w+3; lanes sweep N (coalesced float4 reads).
// Each thread packs a 4k x 4n block into 4 x ds_store_b64.
__device__ __forceinline__ void stage_b_tile(unsigned short* ldsB,
                                             const float* __restrict__ wrow,
                                             int ldN, int tid) {
  const int kk = (tid >> 5) * 4;       // 8 waves * 4 rows = 32 K rows
  const int nn = (tid & 31) * 4;       // 32 lanes * 4 cols = 128 N cols
  float4 v[4];
#pragma unroll
  for (int j = 0; j < 4; ++j)
    v[j] = *(const float4*)(wrow + (size_t)(kk + j) * ldN + nn);
  const float c0[4] = {v[0].x, v[0].y, v[0].z, v[0].w};
  const float c1[4] = {v[1].x, v[1].y, v[1].z, v[1].w};
  const float c2[4] = {v[2].x, v[2].y, v[2].z, v[2].w};
  const float c3[4] = {v[3].x, v[3].y, v[3].z, v[3].w};
#pragma unroll
  for (int i = 0; i < 4; ++i) {
    const unsigned long long q =
        (unsigned long long)pack2(c0[i], c1[i]) |
        ((unsigned long long)pack2(c2[i], c3[i]) << 32);
    *(unsigned long long*)(ldsB + (nn + i) * LSTR + kk) = q;  // 8B aligned
  }
}

// ---------- kernel 1: router (logits, softmax, argmax, per-expert lists) ----------
__global__ __launch_bounds__(256) void router_kernel(
    const float* __restrict__ hidden, const float* __restrict__ rw,
    float* __restrict__ logits_out, int* __restrict__ index_out,
    float* __restrict__ prob_out, int* __restrict__ counts, int* __restrict__ lists)
{
  const int gtid = blockIdx.x * blockDim.x + threadIdx.x;
  const int tok  = gtid >> 5;
  const int lane = gtid & 31;
  if (tok >= NTOK) return;

  float acc[NE];
#pragma unroll
  for (int e = 0; e < NE; ++e) acc[e] = 0.f;

  const float* hrow = hidden + (size_t)tok * DM;
  for (int d = lane; d < DM; d += 32) {
    const float x = hrow[d];
#pragma unroll
    for (int e = 0; e < NE; ++e) acc[e] += x * rw[d * NE + e];
  }
#pragma unroll
  for (int off = 16; off > 0; off >>= 1)
#pragma unroll
    for (int e = 0; e < NE; ++e) acc[e] += __shfl_xor(acc[e], off, 32);

  if (lane == 0) {
    float mx = acc[0]; int am = 0;
#pragma unroll
    for (int e = 1; e < NE; ++e) if (acc[e] > mx) { mx = acc[e]; am = e; }
    float s = 0.f;
#pragma unroll
    for (int e = 0; e < NE; ++e) s += __expf(acc[e] - mx);
#pragma unroll
    for (int e = 0; e < NE; ++e) logits_out[tok * NE + e] = acc[e];
    index_out[tok] = am;
    prob_out[tok]  = 1.f / s;                 // softmax prob of argmax
    const int pos = atomicAdd(&counts[am], 1);
    lists[am * NTOK + pos] = tok;
  }
}

// ---------- kernel 2: h = relu(x @ wi[e]), h stored as bf16 ----------
__global__ __launch_bounds__(256) void moe_ffn1(
    const float* __restrict__ hidden, const float* __restrict__ wi,
    const int* __restrict__ counts, const int* __restrict__ lists,
    unsigned short* __restrict__ hbuf)
{
  const int e  = blockIdx.z;
  const int m0 = blockIdx.y * 128;
  const int n0 = blockIdx.x * 128;
  const int cnt = counts[e];
  if (m0 >= cnt) return;

  const int tid  = threadIdx.x;
  const int lane = tid & 31;
  const int wave = tid >> 5;
  const int wm   = wave >> 1;          // 4 wave-rows  -> 32 M rows each
  const int wn   = wave & 1;           // 2 wave-cols  -> 64 N cols each

  __shared__ __align__(16) unsigned short ldsA[128 * LSTR];
  __shared__ __align__(16) unsigned short ldsB[128 * LSTR];
  __shared__ int tokS[128];

  if (tid < 128) {
    const int m = m0 + tid;
    tokS[tid] = (m < cnt) ? lists[e * NTOK + m] : -1;
  }
  __syncthreads();

  v8f acc[2][4];
#pragma unroll
  for (int s = 0; s < 2; ++s)
#pragma unroll
    for (int t = 0; t < 4; ++t)
#pragma unroll
      for (int j = 0; j < 8; ++j) acc[s][t][j] = 0.f;

  const float* wbase = wi + (size_t)e * DM * DF;

  for (int k0 = 0; k0 < DM; k0 += 32) {
    if (k0 + 32 < DM) __builtin_prefetch(wbase + (size_t)(k0 + 32) * DF + n0, 0, 1);

    // stage A (gathered token rows, fp32 -> bf16): 2 threads/row, 16 floats each
    {
      const int r = tid >> 1;
      const int c = (tid & 1) * 16;
      const int tok = tokS[r];
      unsigned pack[8];
      if (tok >= 0) {
        const float4* s4 = (const float4*)(hidden + (size_t)tok * DM + k0 + c);
#pragma unroll
        for (int j = 0; j < 4; ++j) {
          const float4 v = s4[j];
          pack[2 * j]     = pack2(v.x, v.y);
          pack[2 * j + 1] = pack2(v.z, v.w);
        }
      } else {
#pragma unroll
        for (int j = 0; j < 8; ++j) pack[j] = 0u;
      }
      unsigned* dst = (unsigned*)(ldsA + r * LSTR + c);
#pragma unroll
      for (int j = 0; j < 8; ++j) dst[j] = pack[j];
    }
    // stage B transposed (wi is K x N row-major -> LDS [n][k])
    stage_b_tile(ldsB, wbase + (size_t)k0 * DF + n0, DF, tid);
    __syncthreads();

    v16bf afrag[2];
#pragma unroll
    for (int s = 0; s < 2; ++s)
      afrag[s] = load_frag(ldsA + (wm * 32 + s * 16) * LSTR, lane);
#pragma unroll
    for (int t = 0; t < 4; ++t) {
      const v16bf bfrag = load_frag(ldsB + (wn * 64 + t * 16) * LSTR, lane);
#pragma unroll
      for (int s = 0; s < 2; ++s)
        acc[s][t] = __builtin_amdgcn_wmma_f32_16x16x32_bf16(
            false, afrag[s], false, bfrag, (short)0, acc[s][t], false, false);
    }
    __syncthreads();
  }

  // relu + bf16 store into hbuf[token][ff]
#pragma unroll
  for (int s = 0; s < 2; ++s)
#pragma unroll
    for (int t = 0; t < 4; ++t)
#pragma unroll
      for (int v = 0; v < 8; ++v) {
        const int row = wm * 32 + s * 16 + v + ((lane >> 4) << 3);
        const int col = n0 + wn * 64 + t * 16 + (lane & 15);
        const int tok = tokS[row];
        if (tok >= 0) {
          const float x = acc[s][t][v];
          hbuf[(size_t)tok * DF + col] = f2bf(x > 0.f ? x : 0.f);
        }
      }
}

// ---------- kernel 3: out = prob * (h @ wo[e]) ----------
__global__ __launch_bounds__(256) void moe_ffn2(
    const unsigned short* __restrict__ hbuf, const float* __restrict__ wo,
    const int* __restrict__ counts, const int* __restrict__ lists,
    const float* __restrict__ prob, float* __restrict__ out)
{
  const int e  = blockIdx.z;
  const int m0 = blockIdx.y * 128;
  const int n0 = blockIdx.x * 128;
  const int cnt = counts[e];
  if (m0 >= cnt) return;

  const int tid  = threadIdx.x;
  const int lane = tid & 31;
  const int wave = tid >> 5;
  const int wm   = wave >> 1;
  const int wn   = wave & 1;

  __shared__ __align__(16) unsigned short ldsA[128 * LSTR];
  __shared__ __align__(16) unsigned short ldsB[128 * LSTR];
  __shared__ int   tokS[128];
  __shared__ float probS[128];

  if (tid < 128) {
    const int m = m0 + tid;
    const int tok = (m < cnt) ? lists[e * NTOK + m] : -1;
    tokS[tid]  = tok;
    probS[tid] = (tok >= 0) ? prob[tok] : 0.f;
  }
  __syncthreads();

  v8f acc[2][4];
#pragma unroll
  for (int s = 0; s < 2; ++s)
#pragma unroll
    for (int t = 0; t < 4; ++t)
#pragma unroll
      for (int j = 0; j < 8; ++j) acc[s][t][j] = 0.f;

  const float* wbase = wo + (size_t)e * DF * DM;

  for (int k0 = 0; k0 < DF; k0 += 32) {
    if (k0 + 32 < DF) __builtin_prefetch(wbase + (size_t)(k0 + 32) * DM + n0, 0, 1);

    // stage A: h is already bf16 in global -> async copy straight into LDS
    {
      const int r = tid >> 1;
      const int sb = (tid & 1) * 4;                    // 4 x b64 segments per thread
      const int tok = tokS[r];
      const unsigned short* srow = hbuf + (size_t)(tok < 0 ? 0 : tok) * DF + k0;
#pragma unroll
      for (int s = 0; s < 4; ++s) {
        const unsigned long long ga =
            (unsigned long long)(uintptr_t)(srow + (sb + s) * 4);
        const unsigned la =
            (unsigned)(uintptr_t)(ldsA + r * LSTR + (sb + s) * 4);
        asm volatile("global_load_async_to_lds_b64 %0, %1, off"
                     :: "v"(la), "v"(ga) : "memory");
      }
    }
    // stage B transposed (wo is K x N row-major -> LDS [n][k])
    stage_b_tile(ldsB, wbase + (size_t)k0 * DM + n0, DM, tid);

    asm volatile("s_wait_asynccnt 0" ::: "memory");
    __syncthreads();

    v16bf afrag[2];
#pragma unroll
    for (int s = 0; s < 2; ++s)
      afrag[s] = load_frag(ldsA + (wm * 32 + s * 16) * LSTR, lane);
#pragma unroll
    for (int t = 0; t < 4; ++t) {
      const v16bf bfrag = load_frag(ldsB + (wn * 64 + t * 16) * LSTR, lane);
#pragma unroll
      for (int s = 0; s < 2; ++s)
        acc[s][t] = __builtin_amdgcn_wmma_f32_16x16x32_bf16(
            false, afrag[s], false, bfrag, (short)0, acc[s][t], false, false);
    }
    __syncthreads();
  }

  // scale by router prob, scatter fp32 rows
#pragma unroll
  for (int s = 0; s < 2; ++s)
#pragma unroll
    for (int t = 0; t < 4; ++t)
#pragma unroll
      for (int v = 0; v < 8; ++v) {
        const int row = wm * 32 + s * 16 + v + ((lane >> 4) << 3);
        const int col = n0 + wn * 64 + t * 16 + (lane & 15);
        const int tok = tokS[row];
        if (tok >= 0) out[(size_t)tok * DM + col] = probS[row] * acc[s][t][v];
      }
}

// ---------- launch ----------
extern "C" void kernel_launch(void* const* d_in, const int* in_sizes, int n_in,
                              void* d_out, int out_size, void* d_ws, size_t ws_size,
                              hipStream_t stream) {
  (void)in_sizes; (void)n_in; (void)out_size; (void)ws_size;

  const float* hidden = (const float*)d_in[0];   // [4,1024,768]
  const float* rw     = (const float*)d_in[1];   // [768,8]
  const float* wi     = (const float*)d_in[2];   // [8,768,3072]
  const float* wo     = (const float*)d_in[3];   // [8,3072,768]

  float* out    = (float*)d_out;                       // [4096,768]
  float* logits = out + (size_t)NTOK * DM;             // [4096,8]
  int*   eidx   = (int*)(logits + (size_t)NTOK * NE);  // [4096]

  char* ws = (char*)d_ws;
  int*            counts = (int*)ws;                                    // 64 B
  int*            lists  = (int*)(ws + 64);                             // 8*4096*4
  float*          prob   = (float*)(ws + 64 + NE * NTOK * 4);           // 4096*4
  unsigned short* hbuf   = (unsigned short*)(ws + 64 + NE * NTOK * 4 + NTOK * 4); // 4096*3072*2

  hipMemsetAsync(counts, 0, NE * sizeof(int), stream);

  router_kernel<<<(NTOK * 32) / 256, 256, 0, stream>>>(
      hidden, rw, logits, eidx, prob, counts, lists);

  moe_ffn1<<<dim3(DF / 128, NTOK / 128, NE), 256, 0, stream>>>(
      hidden, wi, counts, lists, hbuf);

  moe_ffn2<<<dim3(DM / 128, NTOK / 128, NE), 256, 0, stream>>>(
      hbuf, wo, counts, lists, prob, out);
}
